// PTBlock_20856361190102
// MI455X (gfx1250) — matrix operand order repaired and checked by reading, not compiled, not run on hardware
//
#include <hip/hip_runtime.h>

#define BATCH 4
#define CH    128
#define NPTS  4096
#define KNN   16

typedef __bf16 v16bf __attribute__((ext_vector_type(16)));
typedef __bf16 v8bf  __attribute__((ext_vector_type(8)));
typedef float  v8f   __attribute__((ext_vector_type(8)));
typedef float  f4    __attribute__((ext_vector_type(4)));

__device__ __forceinline__ v8f vzero8() {
    v8f z;
#pragma unroll
    for (int e = 0; e < 8; ++e) z[e] = 0.0f;
    return z;
}

__device__ __forceinline__ v16bf cat8(v8bf lo, v8bf hi) {
    return __builtin_shufflevector(lo, hi, 0, 1, 2, 3, 4, 5, 6, 7,
                                           8, 9, 10, 11, 12, 13, 14, 15);
}

// ---------------------------------------------------------------------------
// 128x64x128 bf16 WMMA GEMM. Weights row-major (K-contiguous) in Wlds;
// activations K-major (transposed) in Inlds so every fragment is two
// aligned 16-byte LDS vector loads.
// ---------------------------------------------------------------------------
__device__ __forceinline__ void wmma_gemm_128x64(const __bf16 (&Wlds)[CH][136],
                                                 const __bf16 (&Inlds)[64][136],
                                                 int lane, int w, v8f acc[4]) {
    const int mrow   = 16 * w + (lane & 15);
    const int khalfA = (lane >> 4) << 3;   // A: elements 0-7 -> k+0..7, 8-15 -> k+16..23
    const int khalfB = (lane >> 4) << 3;   // B: elements 0-7 -> k+0..7, 8-15 -> k+16..23
    const int ncol0  = lane & 15;
#pragma unroll
    for (int ks = 0; ks < 4; ++ks) {
        const __bf16* pa = &Wlds[mrow][ks * 32 + khalfA];
        const v16bf a = cat8(*(const v8bf*)pa, *(const v8bf*)(pa + 16));
        const int kb = ks * 32 + khalfB;
#pragma unroll
        for (int t = 0; t < 4; ++t) {
            const __bf16* pb = &Inlds[t * 16 + ncol0][kb];
            const v16bf bf = cat8(*(const v8bf*)pb, *(const v8bf*)(pb + 16));
            acc[t] = __builtin_amdgcn_wmma_f32_16x16x32_bf16(
                false, a, false, bf, (short)0, acc[t], false, false);
        }
    }
}

__device__ __forceinline__ void load_weights(const __bf16* __restrict__ Wb,
                                             __bf16 (&Ws)[CH][136], int tid) {
    const v8bf* src = (const v8bf*)Wb;
    for (int i = tid; i < CH * CH / 8; i += 256) {
        const int e0 = i << 3;
        *(v8bf*)&Ws[e0 >> 7][e0 & 127] = src[i];
    }
}

// ---------------------------------------------------------------------------
// Weight f32 -> bf16 conversion
// ---------------------------------------------------------------------------
__global__ void cvt_bf16_kernel(const float* __restrict__ src,
                                __bf16* __restrict__ dst, int n) {
    int i = blockIdx.x * blockDim.x + threadIdx.x;
    if (i < n) dst[i] = (__bf16)src[i];
}

// ---------------------------------------------------------------------------
// KNN: one thread per query point; tiled scan through LDS, register
// insertion network keeps the 16 smallest distances.
// ---------------------------------------------------------------------------
__global__ __launch_bounds__(256)
void knn_kernel(const float* __restrict__ P, int* __restrict__ Idx) {
    __shared__ float sx[256], sy[256], sz[256];
    const int b = blockIdx.y;
    const int n = blockIdx.x * 256 + threadIdx.x;
    const float* Pb = P + (size_t)b * 3 * NPTS;
    const float qx = Pb[n], qy = Pb[NPTS + n], qz = Pb[2 * NPTS + n];

    float bestd[KNN];
    int   besti[KNN];
#pragma unroll
    for (int s = 0; s < KNN; ++s) { bestd[s] = 3.0e38f; besti[s] = 0; }

    for (int m0 = 0; m0 < NPTS; m0 += 256) {
        __syncthreads();
        sx[threadIdx.x] = Pb[m0 + threadIdx.x];
        sy[threadIdx.x] = Pb[NPTS + m0 + threadIdx.x];
        sz[threadIdx.x] = Pb[2 * NPTS + m0 + threadIdx.x];
        __syncthreads();
        for (int j = 0; j < 256; ++j) {
            const float dx = qx - sx[j], dy = qy - sy[j], dz = qz - sz[j];
            const float d2 = dx * dx + dy * dy + dz * dz;
            if (d2 < bestd[KNN - 1]) {
                float d = d2; int id = m0 + j;
#pragma unroll
                for (int s = 0; s < KNN; ++s) {
                    const bool sm = d < bestd[s];
                    const float td = bestd[s]; const int ti = besti[s];
                    bestd[s] = sm ? d : td;  besti[s] = sm ? id : ti;
                    d = sm ? td : d;         id = sm ? ti : id;
                }
            }
        }
    }
    int* out = Idx + ((size_t)b * NPTS + n) * KNN;
#pragma unroll
    for (int s = 0; s < KNN; ++s) out[s] = besti[s];
}

// ---------------------------------------------------------------------------
// Generic affine GEMM: Y[b] = W(128x128) @ X[b] + bias (+ optional residual).
// ---------------------------------------------------------------------------
__global__ __launch_bounds__(256)
void gemm128_kernel(const __bf16* __restrict__ Wb, const float* __restrict__ bias,
                    const float* __restrict__ X, const float* __restrict__ Res,
                    float* __restrict__ Y) {
    __shared__ __bf16 Ws[CH][136];
    __shared__ __bf16 Xt[64][136];   // transposed: Xt[col][k]
    const int tid = threadIdx.x;
    const int lane = tid & 31;
    const int w = tid >> 5;
    const int b = blockIdx.y;
    const int n0 = blockIdx.x * 64;
    const size_t bOff = (size_t)b * CH * NPTS;
    const float* Xb = X + bOff;
    float* Yb = Y + bOff;

    load_weights(Wb, Ws, tid);
    for (int i = tid; i < CH * 16; i += 256) {   // 128 rows x 16 float4
        const int c = i >> 4, n4 = (i & 15) << 2;
        const f4 v = *(const f4*)&Xb[(size_t)c * NPTS + n0 + n4];
        Xt[n4 + 0][c] = (__bf16)v[0];
        Xt[n4 + 1][c] = (__bf16)v[1];
        Xt[n4 + 2][c] = (__bf16)v[2];
        Xt[n4 + 3][c] = (__bf16)v[3];
    }
    __syncthreads();

    v8f acc[4] = {vzero8(), vzero8(), vzero8(), vzero8()};
    wmma_gemm_128x64(Ws, Xt, lane, w, acc);

    const int mbase = 16 * w + ((lane >> 4) << 3);
    const int ncol0 = lane & 15;
#pragma unroll
    for (int t = 0; t < 4; ++t) {
        const int nOut = n0 + 16 * t + ncol0;
#pragma unroll
        for (int v = 0; v < 8; ++v) {
            const int m = mbase + v;
            float val = acc[t][v] + bias[m];
            if (Res) val += Res[bOff + (size_t)m * NPTS + nOut];
            Yb[(size_t)m * NPTS + nOut] = val;
        }
    }
}

// ---------------------------------------------------------------------------
// Fused neighborhood attention: per block = 4 query points (64 neighbor cols).
// d1(VALU) -> d2(WMMA) -> gin -> g1(WMMA,relu) -> g2(WMMA) -> softmax(K) ->
// weighted sum with (alpha_gathered + pos).
// ---------------------------------------------------------------------------
__global__ __launch_bounds__(256)
void attn_kernel(const float* __restrict__ P, const int* __restrict__ Idx,
                 const float* __restrict__ Phi, const float* __restrict__ Psi,
                 const float* __restrict__ Alp,
                 const float* __restrict__ d1W, const float* __restrict__ d1b,
                 const __bf16* __restrict__ d2Wb, const float* __restrict__ d2b,
                 const __bf16* __restrict__ g1Wb, const float* __restrict__ g1b,
                 const __bf16* __restrict__ g2Wb, const float* __restrict__ g2b,
                 float* __restrict__ Ybuf) {
    __shared__ __bf16 Ws[CH][136];    // staged weight (d2 -> g1 -> g2)
    __shared__ __bf16 Ht[64][136];    // transposed bf16 GEMM input (col, k)
    __shared__ float  POS[CH][68];    // d2 output, reused at the end
    __shared__ float  ATT[CH][68];    // g1/g2 outputs, softmax in place
    __shared__ float  relxyz[3][64];
    __shared__ float  phis[CH][4];
    __shared__ int    sidx[64];

    const int tid = threadIdx.x;
    const int lane = tid & 31;
    const int w = tid >> 5;
    const int b = blockIdx.y;
    const int n0 = blockIdx.x * 4;
    const size_t bOff = (size_t)b * CH * NPTS;
    const float* Pb = P + (size_t)b * 3 * NPTS;

    if (tid < 64) {
        const int q = tid >> 4, k = tid & 15;
        sidx[tid] = Idx[((size_t)b * NPTS + n0 + q) * KNN + k];
    }
    __syncthreads();
    for (int i = tid; i < 3 * 64; i += 256) {
        const int t = i >> 6, j = i & 63, q = j >> 4;
        relxyz[t][j] = Pb[t * NPTS + n0 + q] - Pb[t * NPTS + sidx[j]];
    }
    for (int i = tid; i < CH * 4; i += 256) {
        const int c = i >> 2, q = i & 3;
        phis[c][q] = Phi[bOff + (size_t)c * NPTS + n0 + q];
    }
    load_weights(d2Wb, Ws, tid);
    __syncthreads();

    // h1 = relu(d1W @ rel + d1b): inner dim 3 -> plain VALU, stored transposed
    for (int i = tid; i < CH * 64; i += 256) {
        const int c = i >> 6, j = i & 63;
        float v = d1b[c] + d1W[c * 3 + 0] * relxyz[0][j]
                        + d1W[c * 3 + 1] * relxyz[1][j]
                        + d1W[c * 3 + 2] * relxyz[2][j];
        Ht[j][c] = (__bf16)fmaxf(v, 0.0f);
    }
    __syncthreads();

    // pos = d2W @ h1 + d2b
    {
        v8f acc[4] = {vzero8(), vzero8(), vzero8(), vzero8()};
        wmma_gemm_128x64(Ws, Ht, lane, w, acc);
        const int mbase = 16 * w + ((lane >> 4) << 3);
        const int ncol0 = lane & 15;
#pragma unroll
        for (int t = 0; t < 4; ++t)
#pragma unroll
            for (int v = 0; v < 8; ++v)
                POS[mbase + v][16 * t + ncol0] = acc[t][v] + d2b[mbase + v];
    }
    __syncthreads();

    // gin = phi - psi_gathered + pos ; stage g1 weights
    for (int i = tid; i < CH * 64; i += 256) {
        const int c = i >> 6, j = i & 63, q = j >> 4;
        const float v = phis[c][q] - Psi[bOff + (size_t)c * NPTS + sidx[j]] + POS[c][j];
        Ht[j][c] = (__bf16)v;
    }
    load_weights(g1Wb, Ws, tid);
    __syncthreads();

    // t1 = relu(g1W @ gin + g1b)
    {
        v8f acc[4] = {vzero8(), vzero8(), vzero8(), vzero8()};
        wmma_gemm_128x64(Ws, Ht, lane, w, acc);
        const int mbase = 16 * w + ((lane >> 4) << 3);
        const int ncol0 = lane & 15;
#pragma unroll
        for (int t = 0; t < 4; ++t)
#pragma unroll
            for (int v = 0; v < 8; ++v)
                ATT[mbase + v][16 * t + ncol0] =
                    fmaxf(acc[t][v] + g1b[mbase + v], 0.0f);
    }
    __syncthreads();

    for (int i = tid; i < CH * 64; i += 256) {
        const int c = i >> 6, j = i & 63;
        Ht[j][c] = (__bf16)ATT[c][j];
    }
    load_weights(g2Wb, Ws, tid);
    __syncthreads();

    // attn = g2W @ t1 + g2b
    {
        v8f acc[4] = {vzero8(), vzero8(), vzero8(), vzero8()};
        wmma_gemm_128x64(Ws, Ht, lane, w, acc);
        const int mbase = 16 * w + ((lane >> 4) << 3);
        const int ncol0 = lane & 15;
#pragma unroll
        for (int t = 0; t < 4; ++t)
#pragma unroll
            for (int v = 0; v < 8; ++v)
                ATT[mbase + v][16 * t + ncol0] = acc[t][v] + g2b[mbase + v];
    }
    __syncthreads();

    // softmax over K=16 and weighted reduction with (alpha_gathered + pos)
    for (int i = tid; i < CH * 4; i += 256) {
        const int c = i >> 2, q = i & 3;
        float mx = -3.0e38f;
#pragma unroll
        for (int k = 0; k < KNN; ++k) mx = fmaxf(mx, ATT[c][q * 16 + k]);
        float ek[KNN]; float s = 0.0f;
#pragma unroll
        for (int k = 0; k < KNN; ++k) { ek[k] = __expf(ATT[c][q * 16 + k] - mx); s += ek[k]; }
        const float inv = 1.0f / s;
        float y = 0.0f;
#pragma unroll
        for (int k = 0; k < KNN; ++k) {
            const int j = q * 16 + k;
            const float av = Alp[bOff + (size_t)c * NPTS + sidx[j]] + POS[c][j];
            y += ek[k] * inv * av;
        }
        Ybuf[bOff + (size_t)c * NPTS + n0 + q] = y;
    }
}

// ---------------------------------------------------------------------------
extern "C" void kernel_launch(void* const* d_in, const int* in_sizes, int n_in,
                              void* d_out, int out_size, void* d_ws, size_t ws_size,
                              hipStream_t stream) {
    const float* input_p = (const float*)d_in[0];
    const float* input_x = (const float*)d_in[1];
    const float* top_W   = (const float*)d_in[2];  const float* top_b   = (const float*)d_in[3];
    const float* down_W  = (const float*)d_in[4];  const float* down_b  = (const float*)d_in[5];
    const float* phi_W   = (const float*)d_in[6];  const float* phi_b   = (const float*)d_in[7];
    const float* psi_W   = (const float*)d_in[8];  const float* psi_b   = (const float*)d_in[9];
    const float* alpha_W = (const float*)d_in[10]; const float* alpha_b = (const float*)d_in[11];
    const float* g1_W    = (const float*)d_in[12]; const float* g1_b    = (const float*)d_in[13];
    const float* g2_W    = (const float*)d_in[14]; const float* g2_b    = (const float*)d_in[15];
    const float* d1_W    = (const float*)d_in[16]; const float* d1_b    = (const float*)d_in[17];
    const float* d2_W    = (const float*)d_in[18]; const float* d2_b    = (const float*)d_in[19];

    char* ws = (char*)d_ws;
    size_t off = 0;
    auto alloc = [&](size_t bytes) -> void* {
        void* p = ws + off;
        off = (off + bytes + 255) & ~(size_t)255;
        return p;
    };
    const size_t WSZ = (size_t)CH * CH;           // 16384 elems
    const size_t FSZ = (size_t)BATCH * CH * NPTS; // 2M elems

    __bf16* wTop  = (__bf16*)alloc(WSZ * 2);
    __bf16* wPhi  = (__bf16*)alloc(WSZ * 2);
    __bf16* wPsi  = (__bf16*)alloc(WSZ * 2);
    __bf16* wAlp  = (__bf16*)alloc(WSZ * 2);
    __bf16* wD2   = (__bf16*)alloc(WSZ * 2);
    __bf16* wG1   = (__bf16*)alloc(WSZ * 2);
    __bf16* wG2   = (__bf16*)alloc(WSZ * 2);
    __bf16* wDown = (__bf16*)alloc(WSZ * 2);
    int*   idxbuf = (int*)  alloc((size_t)BATCH * NPTS * KNN * 4);
    float* xbuf   = (float*)alloc(FSZ * 4);
    float* phibuf = (float*)alloc(FSZ * 4);
    float* psibuf = (float*)alloc(FSZ * 4);
    float* alpbuf = (float*)alloc(FSZ * 4);
    float* ybuf   = (float*)alloc(FSZ * 4);

    const int cvtGrid = (int)(WSZ / 256);
    cvt_bf16_kernel<<<cvtGrid, 256, 0, stream>>>(top_W,   wTop,  (int)WSZ);
    cvt_bf16_kernel<<<cvtGrid, 256, 0, stream>>>(phi_W,   wPhi,  (int)WSZ);
    cvt_bf16_kernel<<<cvtGrid, 256, 0, stream>>>(psi_W,   wPsi,  (int)WSZ);
    cvt_bf16_kernel<<<cvtGrid, 256, 0, stream>>>(alpha_W, wAlp,  (int)WSZ);
    cvt_bf16_kernel<<<cvtGrid, 256, 0, stream>>>(d2_W,    wD2,   (int)WSZ);
    cvt_bf16_kernel<<<cvtGrid, 256, 0, stream>>>(g1_W,    wG1,   (int)WSZ);
    cvt_bf16_kernel<<<cvtGrid, 256, 0, stream>>>(g2_W,    wG2,   (int)WSZ);
    cvt_bf16_kernel<<<cvtGrid, 256, 0, stream>>>(down_W,  wDown, (int)WSZ);

    knn_kernel<<<dim3(NPTS / 256, BATCH), 256, 0, stream>>>(input_p, idxbuf);

    gemm128_kernel<<<dim3(NPTS / 64, BATCH), 256, 0, stream>>>(wTop, top_b,   input_x, nullptr, xbuf);
    gemm128_kernel<<<dim3(NPTS / 64, BATCH), 256, 0, stream>>>(wPhi, phi_b,   xbuf,    nullptr, phibuf);
    gemm128_kernel<<<dim3(NPTS / 64, BATCH), 256, 0, stream>>>(wPsi, psi_b,   xbuf,    nullptr, psibuf);
    gemm128_kernel<<<dim3(NPTS / 64, BATCH), 256, 0, stream>>>(wAlp, alpha_b, xbuf,    nullptr, alpbuf);

    attn_kernel<<<dim3(NPTS / 4, BATCH), 256, 0, stream>>>(
        input_p, idxbuf, phibuf, psibuf, alpbuf,
        d1_W, d1_b, wD2, d2_b, wG1, g1_b, wG2, g2_b, ybuf);

    gemm128_kernel<<<dim3(NPTS / 64, BATCH), 256, 0, stream>>>(wDown, down_b, ybuf, input_x, (float*)d_out);

    (void)in_sizes; (void)n_in; (void)out_size; (void)ws_size;
}
